// Generator_34746285425383
// MI455X (gfx1250) — compile-verified, matrix operand order checked by reading, not run on hardware
//
#include <hip/hip_runtime.h>
#include <hip/hip_bf16.h>

typedef __attribute__((ext_vector_type(16))) _Float16 v16h;
typedef __attribute__((ext_vector_type(8)))  _Float16 h8;
typedef __attribute__((ext_vector_type(8)))  float    v8f;

#define H        256
#define H3       768
#define LATENT   128
#define MAXN     64
#define END_NODE 11
#define NTHREADS 512
#define TILES_PER_WAVE 3          // 48 N-tiles / 16 waves
#define W16_ELEMS (4 * H3 * H)    // nWih,nWhh,eWih,eWhh as f16

// ---------------- output layout (floats, concatenated in return order) ----
// node_probs (64,3)      @ 0
// node_idxs  (64,3)      @ 192
// node_valid (64,)       @ 384
// edge_probs (64,64,2)   @ 448
// edge_idxs  (64,64,2)   @ 8640
// edge_valid (64,64)     @ 16832   -> total 20928

__device__ __forceinline__ float sigm(float x) { return 1.f / (1.f + __expf(-x)); }

__device__ __forceinline__ v16h cat8(h8 lo, h8 hi) {
    return __builtin_shufflevector(lo, hi, 0, 1, 2, 3, 4, 5, 6, 7,
                                           8, 9, 10, 11, 12, 13, 14, 15);
}

// Dual mat-vec via one shared A tile:
//   A (16x256 f16, LDS): row0 = x, row1 = h, rows 2..15 = 0
//   gi[n] = sum_k x[k] * Wi[n][k]   (D row M=0 -> c[0], lanes 0-15)
//   gh[n] = sum_k h[k] * Wh[n][k]   (D row M=1 -> c[1], lanes 0-15)
// 16 waves x 3 N-tiles x 8 K-tiles; fragments are 2x16B contiguous loads.
__device__ __forceinline__ void gru_gemv2(const _Float16* __restrict__ Wi,
                                          const _Float16* __restrict__ Wh,
                                          const _Float16* __restrict__ shA,
                                          float* __restrict__ gi,
                                          float* __restrict__ gh,
                                          int wave, int lane)
{
    const int m  = lane & 15;
    const int hi = lane >> 4;
    const int cb = hi * 8;                      // K sub-offset for this lane half
    const _Float16* arow = shA + m * H + cb;
#pragma unroll
    for (int tt = 0; tt < TILES_PER_WAVE; ++tt) {
        const int nbase = (wave * TILES_PER_WAVE + tt) * 16;
        const int n     = nbase + m;            // B column for this lane
        const _Float16* wi = Wi + n * H + cb;
        const _Float16* wh = Wh + n * H + cb;
        v8f ci = {}, cg = {};
#pragma unroll 2
        for (int kt = 0; kt < 8; ++kt) {
            const int k0 = kt * 32;
            v16h a  = cat8(*(const h8*)(arow + k0), *(const h8*)(arow + k0 + 16));
            v16h bi = cat8(*(const h8*)(wi + k0),   *(const h8*)(wi + k0 + 16));
            v16h bh = cat8(*(const h8*)(wh + k0),   *(const h8*)(wh + k0 + 16));
            ci = __builtin_amdgcn_wmma_f32_16x16x32_f16(false, a, false, bi,
                                                        (short)0, ci, false, false);
            cg = __builtin_amdgcn_wmma_f32_16x16x32_f16(false, a, false, bh,
                                                        (short)0, cg, false, false);
        }
        if (hi == 0) {
            gi[nbase + m] = ci[0];              // M=0 row of D
            gh[nbase + m] = cg[1];              // M=1 row of D
        }
    }
}

// GRU elementwise combine (per-thread, own index only: race-free in place)
__device__ __forceinline__ void gru_combine(const float* __restrict__ bih,
                                            const float* __restrict__ bhh,
                                            const float* __restrict__ gi,
                                            const float* __restrict__ gh,
                                            const float* __restrict__ hprev,
                                            float* __restrict__ hout, int tid)
{
    if (tid < H) {
        float r  = sigm(gi[tid]          + bih[tid]         + gh[tid]         + bhh[tid]);
        float zg = sigm(gi[H + tid]      + bih[H + tid]     + gh[H + tid]     + bhh[H + tid]);
        float n  = tanhf(gi[2 * H + tid] + bih[2 * H + tid] +
                         r * (gh[2 * H + tid] + bhh[2 * H + tid]));
        hout[tid] = (1.f - zg) * n + zg * hprev[tid];
    }
    __syncthreads();
}

// Linear(f,256) -> softmax -> (max prob, argmax). All threads must call.
__device__ __forceinline__ void classify(const float* __restrict__ W,
                                         const float* __restrict__ b, int f,
                                         const float* __restrict__ vec,
                                         float* p_out, int* i_out,
                                         float* s_logits, int tid)
{
    if (tid < f) {
        float s = b[tid];
        for (int k = 0; k < H; k += 4) {
            s += W[tid * H + k + 0] * vec[k + 0];
            s += W[tid * H + k + 1] * vec[k + 1];
            s += W[tid * H + k + 2] * vec[k + 2];
            s += W[tid * H + k + 3] * vec[k + 3];
        }
        s_logits[tid] = s;
    }
    __syncthreads();
    if (tid == 0) {
        float mx = s_logits[0]; int am = 0;
        for (int c = 1; c < f; ++c) if (s_logits[c] > mx) { mx = s_logits[c]; am = c; }
        float sum = 0.f;
        for (int c = 0; c < f; ++c) sum += __expf(s_logits[c] - mx);
        *p_out = 1.f / sum;          // softmax max prob = exp(0)/sum
        *i_out = am;
    }
    __syncthreads();
}

__global__ void zero_f32(float* p, int n)
{
    for (int i = blockIdx.x * blockDim.x + threadIdx.x; i < n; i += gridDim.x * blockDim.x)
        p[i] = 0.f;
}

__global__ void convert_weights(const float* __restrict__ nWih, const float* __restrict__ nWhh,
                                const float* __restrict__ eWih, const float* __restrict__ eWhh,
                                _Float16* __restrict__ W16)
{
    const int per = H3 * H;
    for (int idx = blockIdx.x * blockDim.x + threadIdx.x; idx < 4 * per;
         idx += gridDim.x * blockDim.x) {
        const int which = idx / per, off = idx % per;
        const float* src = (which == 0) ? nWih : (which == 1) ? nWhh
                                               : (which == 2) ? eWih : eWhh;
        W16[idx] = (_Float16)src[off];
    }
}

__global__ void __launch_bounds__(NTHREADS)
generate_kernel(const float* __restrict__ z,
                const float* __restrict__ lp_w, const float* __restrict__ lp_b,
                const float* __restrict__ nbih, const float* __restrict__ nbhh,
                const float* __restrict__ ebih, const float* __restrict__ ebhh,
                const float* __restrict__ nc0_w, const float* __restrict__ nc0_b,
                const float* __restrict__ nc1_w, const float* __restrict__ nc1_b,
                const float* __restrict__ nc2_w, const float* __restrict__ nc2_b,
                const float* __restrict__ ec0_w, const float* __restrict__ ec0_b,
                const float* __restrict__ ec1_w, const float* __restrict__ ec1_b,
                const _Float16* __restrict__ W16,
                float* __restrict__ buf,   // (64,256) node embedding history
                float* __restrict__ out)
{
    __shared__ _Float16 shA[16 * H];       // WMMA A tile: row0=x, row1=h, rest 0
    __shared__ float s_gi[H3], s_gh[H3];
    __shared__ float s_h[H];               // node hidden
    __shared__ float s_h1[H];              // candidate node embedding
    __shared__ float s_he[H];              // edge hidden
    __shared__ float s_logits[16];
    __shared__ float s_ep[MAXN][2];
    __shared__ int   s_ei[MAXN][2];
    __shared__ float s_np[3];
    __shared__ int   s_ni[3];
    __shared__ int   s_flags[4];           // 0:done 1:allzero 2:node_ok 3:active

    const int tid  = threadIdx.x;
    const int wave = tid >> 5;
    const int lane = tid & 31;

    const _Float16* nWih16 = W16;
    const _Float16* nWhh16 = W16 + 1 * H3 * H;
    const _Float16* eWih16 = W16 + 2 * H3 * H;
    const _Float16* eWhh16 = W16 + 3 * H3 * H;

    // Warm L2 with the full f16 weight image (global_prefetch_b8 path)
    const char* wbytes = (const char*)W16;
    for (size_t ofs = (size_t)tid * 128; ofs < (size_t)W16_ELEMS * 2;
         ofs += (size_t)NTHREADS * 128)
        __builtin_prefetch(wbytes + ofs, 0, 2);

    // Zero the padded A tile once (rows 2..15 stay zero forever)
    for (int idx = tid; idx < 16 * H; idx += NTHREADS) shA[idx] = (_Float16)0.f;

    // h0 = z @ lp_w.T + lp_b
    if (tid < H) {
        float s = lp_b[tid];
        for (int k = 0; k < LATENT; ++k) s += z[k] * lp_w[tid * LATENT + k];
        s_h[tid] = s;
    }
    if (tid == 0) s_flags[0] = 0;
    __syncthreads();

    for (int i = 0; i < MAXN; ++i) {
        // ---- node GRU step: h1 = GRU(x, h) --------------------------------
        if (tid < H) {
            float xv = (i == 0) ? 0.f : buf[(i - 1) * H + tid];
            shA[tid]     = (_Float16)xv;        // A row 0 = x
            shA[H + tid] = (_Float16)s_h[tid];  // A row 1 = h
        }
        __syncthreads();
        gru_gemv2(nWih16, nWhh16, shA, s_gi, s_gh, wave, lane);
        __syncthreads();
        gru_combine(nbih, nbhh, s_gi, s_gh, s_h, s_h1, tid);

        // ---- node classifiers ---------------------------------------------
        classify(nc0_w, nc0_b, 12, s_h1, &s_np[0], &s_ni[0], s_logits, tid);
        classify(nc1_w, nc1_b, 5,  s_h1, &s_np[1], &s_ni[1], s_logits, tid);
        classify(nc2_w, nc2_b, 4,  s_h1, &s_np[2], &s_ni[2], s_logits, tid);

        // ---- edge scan over previous nodes, most recent first -------------
        if (tid < H) s_he[tid] = s_h1[tid];
        if (tid == 0) s_flags[1] = 1;       // all first-edge-argmax == 0 so far
        __syncthreads();
        for (int t = 0; t < i; ++t) {
            const int src = i - 1 - t;
            if (tid < H) {
                shA[tid]     = (_Float16)buf[src * H + tid];  // A row 0 = x_e
                shA[H + tid] = (_Float16)s_he[tid];           // A row 1 = he
            }
            __syncthreads();
            gru_gemv2(eWih16, eWhh16, shA, s_gi, s_gh, wave, lane);
            __syncthreads();
            gru_combine(ebih, ebhh, s_gi, s_gh, s_he, s_he, tid);

            classify(ec0_w, ec0_b, 5, s_he, &s_ep[src][0], &s_ei[src][0], s_logits, tid);
            classify(ec1_w, ec1_b, 4, s_he, &s_ep[src][1], &s_ei[src][1], s_logits, tid);
            if (tid == 0 && s_ei[src][0] != 0) s_flags[1] = 0;
            __syncthreads();
        }

        // ---- decision logic ------------------------------------------------
        if (tid == 0) {
            const int done     = s_flags[0];
            const int is_end   = (s_ni[0] == END_NODE);
            const int no_edges = (i > 0) && s_flags[1];
            const int active   = !done;
            s_flags[2] = active && !is_end && !no_edges;     // node_ok
            s_flags[3] = active;
            s_flags[0] = done || is_end || no_edges;
        }
        __syncthreads();
        const int node_ok = s_flags[2];
        const int active  = s_flags[3];

        // h_next = node_ok&&i>0 ? he : h1 ; if !active keep h
        if (tid < H) {
            float hn = (node_ok && i > 0) ? s_he[tid] : s_h1[tid];
            if (!active) hn = s_h[tid];
            s_h[tid] = hn;                      // own-index only: race-free
            if (node_ok) buf[i * H + tid] = s_h1[tid];
        }

        // ---- outputs -------------------------------------------------------
        if (tid < 3) {
            out[i * 3 + tid]       = node_ok ? s_np[tid] : 0.f;
            out[192 + i * 3 + tid] = (float)s_ni[tid];
        }
        if (tid == 0) out[384 + i] = node_ok ? 1.f : 0.f;
        if (tid < i) {                          // j = tid in [0, i)
            const int j      = tid;
            const int evalid = node_ok && (i > 0);
            out[448  + (i * MAXN + j) * 2 + 0] = evalid ? s_ep[j][0] : 0.f;
            out[448  + (i * MAXN + j) * 2 + 1] = evalid ? s_ep[j][1] : 0.f;
            out[8640 + (i * MAXN + j) * 2 + 0] = (float)s_ei[j][0];
            out[8640 + (i * MAXN + j) * 2 + 1] = (float)s_ei[j][1];
            out[16832 + i * MAXN + j]          = evalid ? 1.f : 0.f;
        }
        __syncthreads();
    }
}

extern "C" void kernel_launch(void* const* d_in, const int* in_sizes, int n_in,
                              void* d_out, int out_size, void* d_ws, size_t ws_size,
                              hipStream_t stream) {
    const float* z     = (const float*)d_in[0];
    const float* lp_w  = (const float*)d_in[1];
    const float* lp_b  = (const float*)d_in[2];
    const float* nWih  = (const float*)d_in[3];
    const float* nWhh  = (const float*)d_in[4];
    const float* nbih  = (const float*)d_in[5];
    const float* nbhh  = (const float*)d_in[6];
    const float* eWih  = (const float*)d_in[7];
    const float* eWhh  = (const float*)d_in[8];
    const float* ebih  = (const float*)d_in[9];
    const float* ebhh  = (const float*)d_in[10];
    const float* nc0_w = (const float*)d_in[11];
    const float* nc0_b = (const float*)d_in[12];
    const float* nc1_w = (const float*)d_in[13];
    const float* nc1_b = (const float*)d_in[14];
    const float* nc2_w = (const float*)d_in[15];
    const float* nc2_b = (const float*)d_in[16];
    const float* ec0_w = (const float*)d_in[17];
    const float* ec0_b = (const float*)d_in[18];
    const float* ec1_w = (const float*)d_in[19];
    const float* ec1_b = (const float*)d_in[20];

    _Float16* W16 = (_Float16*)d_ws;
    float*    buf = (float*)((char*)d_ws + (size_t)W16_ELEMS * sizeof(_Float16));
    float*    out = (float*)d_out;

    // d_out and buf are poisoned: zero them (kernel is deterministic per call)
    zero_f32<<<64, 256, 0, stream>>>(out, out_size);
    zero_f32<<<16, 256, 0, stream>>>(buf, MAXN * H);

    // f32 -> f16 weight repack (once per call; cheap)
    convert_weights<<<256, 256, 0, stream>>>(nWih, nWhh, eWih, eWhh, W16);

    // persistent single-WGP generator
    generate_kernel<<<1, NTHREADS, 0, stream>>>(
        z, lp_w, lp_b, nbih, nbhh, ebih, ebhh,
        nc0_w, nc0_b, nc1_w, nc1_b, nc2_w, nc2_b,
        ec0_w, ec0_b, ec1_w, ec1_b,
        W16, buf, out);
}
